// SelfAttention_61658550501938
// MI455X (gfx1250) — matrix operand order repaired and checked by reading, not compiled, not run on hardware
//
#include <hip/hip_runtime.h>
#include <hip/hip_bf16.h>
#include <stdint.h>

// ---------------------------------------------------------------------------
// CDNA5 / gfx1250 causal self-attention (QKV proj + flash attention), bf16 WMMA
//   x      : fp32 [B=4, T=2048, C=1024]
//   w_qkv  : fp32 [C=1024, 3C=3072]
//   out    : fp32 [B, T, C]
// Workspace: Q [B,H,T,D] bf16, K [B,H,T,D] bf16, V^T [B,H,D,T] bf16 (48 MB)
// ---------------------------------------------------------------------------

typedef float  v8f   __attribute__((ext_vector_type(8)));
typedef __bf16 v16bf __attribute__((ext_vector_type(16)));
typedef __bf16 v4bf  __attribute__((ext_vector_type(4)));

union FragBF {
  v16bf v;
  uint4 q[2];
};

__device__ __forceinline__ v8f vzero8() {
  v8f v;
#pragma unroll
  for (int i = 0; i < 8; ++i) v[i] = 0.0f;
  return v;
}

__device__ __forceinline__ v8f wmma_bf16(const FragBF& a, const FragBF& b, v8f c) {
  // D(f32 16x16) = A(bf16 16x32) * B(bf16 32x16) + C
  return __builtin_amdgcn_wmma_f32_16x16x32_bf16(
      /*neg_a=*/false, a.v, /*neg_b=*/false, b.v,
      /*c_mod=*/(short)0, c, /*reuse_a=*/false, /*reuse_b=*/false);
}

// ---------------------------------------------------------------------------
// Kernel 1: qkv = x @ w_qkv, emitted as bf16 Q/K (B,H,T,D) and V^T (B,H,D,T)
// Tile: BM=128 x BN=128 x BK=64. 256 threads = 8 waves (4 M x 2 N),
// each wave computes 32x64 via 2x4 accumulators of 16x16.
// Software-pipelined: next tile's global loads issue before current WMMAs.
// ---------------------------------------------------------------------------
__global__ __launch_bounds__(256) void qkv_gemm_kernel(
    const float* __restrict__ x, const float* __restrict__ wq,
    __bf16* __restrict__ Qg, __bf16* __restrict__ Kg, __bf16* __restrict__ Vtg)
{
  constexpr int LD = 72;  // padded bf16 row stride (144 B, 16B-aligned rows)
  __shared__ __align__(16) __bf16 As[128 * LD];  // [m][k]
  __shared__ __align__(16) __bf16 Bs[128 * LD];  // [n][k]  (B staged transposed)

  const int tid  = threadIdx.x;
  const int lane = tid & 31;
  const int wave = tid >> 5;
  const int wm   = wave >> 1;     // 0..3
  const int wn   = wave & 1;      // 0..1
  const int m0   = blockIdx.y * 128;
  const int n0   = blockIdx.x * 128;

  v8f acc[2][4];
#pragma unroll
  for (int i = 0; i < 2; ++i)
#pragma unroll
    for (int j = 0; j < 4; ++j) acc[i][j] = vzero8();

  const int  mlo = lane & 15;
  const bool lo  = lane < 16;

  // Register staging buffer for one (A,B) tile: 16 float4 per thread.
  float4 xa[8], wb[8];
  auto issue_tile_loads = [&](int k0) {
#pragma unroll
    for (int ii = 0; ii < 8; ++ii) {
      const int idx = tid + 256 * ii;            // 0..2047
      xa[ii] = *(const float4*)(x + (size_t)(m0 + (idx >> 4)) * 1024 + k0 +
                                ((idx & 15) << 2));
    }
#pragma unroll
    for (int ii = 0; ii < 8; ++ii) {
      const int idx = tid + 256 * ii;            // 0..2047
      wb[ii] = *(const float4*)(wq + (size_t)(k0 + (idx >> 5)) * 3072 + n0 +
                                ((idx & 31) << 2));
    }
  };

  issue_tile_loads(0);

  for (int k0 = 0; k0 < 1024; k0 += 64) {
    // ---- stage registers -> LDS (waits drain the batched loads) ----
#pragma unroll
    for (int ii = 0; ii < 8; ++ii) {
      const int idx = tid + 256 * ii;
      const int row = idx >> 4;                  // 0..127
      const int c4  = (idx & 15) << 2;           // 0..60
      v4bf bv;
      bv[0] = (__bf16)xa[ii].x; bv[1] = (__bf16)xa[ii].y;
      bv[2] = (__bf16)xa[ii].z; bv[3] = (__bf16)xa[ii].w;
      *(v4bf*)&As[row * LD + c4] = bv;
    }
#pragma unroll
    for (int ii = 0; ii < 8; ++ii) {
      const int idx = tid + 256 * ii;
      const int kk  = idx >> 5;                  // 0..63
      const int n4  = (idx & 31) << 2;           // 0..124
      Bs[(n4 + 0) * LD + kk] = (__bf16)wb[ii].x;
      Bs[(n4 + 1) * LD + kk] = (__bf16)wb[ii].y;
      Bs[(n4 + 2) * LD + kk] = (__bf16)wb[ii].z;
      Bs[(n4 + 3) * LD + kk] = (__bf16)wb[ii].w;
    }
    __syncthreads();

    // ---- issue next tile's global loads; latency hides under the WMMAs ----
    if (k0 + 64 < 1024) issue_tile_loads(k0 + 64);

    // ---- two WMMA K-steps over this LDS tile ----
#pragma unroll
    for (int s = 0; s < 2; ++s) {
      const int kb    = s * 32;
      const int abase = kb + (lo ? 0 : 16);   // A layout half-wave K split
      const int bbase = kb + (lo ? 0 : 16);   // B layout half-wave K split
      FragBF af[2];
#pragma unroll
      for (int i = 0; i < 2; ++i) {
        const __bf16* p = &As[(wm * 32 + i * 16 + mlo) * LD + abase];
        af[i].q[0] = *(const uint4*)p;          // K = base .. base+7
        af[i].q[1] = *(const uint4*)(p + 8);    // K = base+8 .. base+15
      }
      FragBF bfm[4];
#pragma unroll
      for (int j = 0; j < 4; ++j) {
        const __bf16* p = &Bs[(wn * 64 + j * 16 + mlo) * LD + bbase];
        bfm[j].q[0] = *(const uint4*)p;         // K = base .. base+7
        bfm[j].q[1] = *(const uint4*)(p + 8);   // K = base+8 .. base+15
      }
#pragma unroll
      for (int i = 0; i < 2; ++i)
#pragma unroll
        for (int j = 0; j < 4; ++j)
          acc[i][j] = wmma_bf16(af[i], bfm[j], acc[i][j]);
    }
    __syncthreads();
  }

  // ---- epilogue: scatter bf16 into Q / K / V^T (block column is uniform) ----
  const int section = n0 >> 10;  // 0=Q, 1=K, 2=V
#pragma unroll
  for (int i = 0; i < 2; ++i)
#pragma unroll
    for (int j = 0; j < 4; ++j)
#pragma unroll
      for (int r = 0; r < 8; ++r) {
        int gm = m0 + wm * 32 + i * 16 + (lo ? r : r + 8);
        int gn = n0 + wn * 64 + j * 16 + mlo;
        int b = gm >> 11, t = gm & 2047;
        int c = gn & 1023;
        int h = c >> 6, d = c & 63;
        __bf16 v = (__bf16)acc[i][j][r];
        if (section == 0)
          Qg[(((size_t)(b * 16 + h)) * 2048 + t) * 64 + d] = v;
        else if (section == 1)
          Kg[(((size_t)(b * 16 + h)) * 2048 + t) * 64 + d] = v;
        else
          Vtg[(((size_t)(b * 16 + h)) * 64 + d) * 2048 + t] = v;
      }
}

// ---------------------------------------------------------------------------
// Kernel 2: causal flash attention. Block = 4 waves, each wave owns a 16-row
// query tile; loop over 32-key blocks with online softmax; S and PV via WMMA.
// K and V fragments are batch-loaded at the top of each iteration so vmem
// latency overlaps the S-WMMAs and softmax instead of the PV tail.
// ---------------------------------------------------------------------------
__global__ __launch_bounds__(128) void flash_attn_kernel(
    const __bf16* __restrict__ Qg, const __bf16* __restrict__ Kg,
    const __bf16* __restrict__ Vtg, float* __restrict__ out)
{
  __shared__ __align__(16) __bf16 Plds[4 * 16 * 32];  // per-wave P staging

  const int tid  = threadIdx.x;
  const int lane = tid & 31;
  const int wave = tid >> 5;
  const int b = blockIdx.z, h = blockIdx.y;
  const int q0 = blockIdx.x * 64 + wave * 16;       // this wave's query tile
  const size_t bh = (size_t)(b * 16 + h);
  const __bf16* Qb = Qg  + bh * 2048 * 64;
  const __bf16* Kb = Kg  + bh * 2048 * 64;
  const __bf16* Vb = Vtg + bh * 64 * 2048;
  __bf16* pl = Plds + wave * (16 * 32);

  const int  n     = lane & 15;
  const bool lo    = lane < 16;
  const int  abase = lo ? 0 : 16;
  const int  bbase = lo ? 0 : 16;

  // Q fragments (A layout, 16x32 each over D=64), kept in registers
  FragBF qa[2];
#pragma unroll
  for (int f = 0; f < 2; ++f) {
    const __bf16* p = Qb + (size_t)(q0 + n) * 64 + f * 32 + abase;
    qa[f].q[0] = *(const uint4*)p;
    qa[f].q[1] = *(const uint4*)(p + 8);
  }

  v8f o[4];
#pragma unroll
  for (int j = 0; j < 4; ++j) o[j] = vzero8();
  float mrow[8], lrow[8];
#pragma unroll
  for (int r = 0; r < 8; ++r) { mrow[r] = -3.0e38f; lrow[r] = 0.0f; }

  for (int kb = 0; kb < q0 + 16; kb += 32) {
    // ---- batch-issue all K and V fragment loads for this key block ----
    FragBF kf[2][2];   // [key subtile][D half]
    FragBF vf[4];      // [d subtile]
    const int kkey = kb + (lo ? 0 : 16);
#pragma unroll
    for (int a = 0; a < 2; ++a) {
      const __bf16* krow = Kb + (size_t)(kb + a * 16 + n) * 64;
#pragma unroll
      for (int f = 0; f < 2; ++f) {
        const __bf16* p = krow + f * 32 + bbase;
        kf[a][f].q[0] = *(const uint4*)p;
        kf[a][f].q[1] = *(const uint4*)(p + 8);
      }
    }
#pragma unroll
    for (int j = 0; j < 4; ++j) {
      const __bf16* vrow = Vb + (size_t)(j * 16 + n) * 2048 + kkey;
      vf[j].q[0] = *(const uint4*)vrow;
      vf[j].q[1] = *(const uint4*)(vrow + 8);
    }
    if (kb + 32 < q0 + 16) {  // speculative prefetch of next key block
      __builtin_prefetch(Kb + (size_t)(kb + 32 + n) * 64, 0, 3);
      __builtin_prefetch(Vb + (size_t)n * 2048 + kb + 32, 0, 3);
    }

    // ---- S = Q * K^T for 32 keys (two 16-key subtiles, two D halves) ----
    v8f s0 = vzero8(), s1 = vzero8();
#pragma unroll
    for (int f = 0; f < 2; ++f) {
      s0 = wmma_bf16(qa[f], kf[0][f], s0);
      s1 = wmma_bf16(qa[f], kf[1][f], s1);
    }

    // ---- online softmax (row lives in one 16-lane half; xor-reduce) ----
    const bool domask = (kb + 31 > q0);
#pragma unroll
    for (int r = 0; r < 8; ++r) {
      const int mm = lo ? r : r + 8;
      const int q  = q0 + mm;
      float a0 = s0[r] * 0.125f;           // scale = 1/sqrt(64)
      float a1 = s1[r] * 0.125f;
      if (domask) {
        if (kb + n > q)      a0 = -3.0e38f;
        if (kb + 16 + n > q) a1 = -3.0e38f;
      }
      float rmax = fmaxf(a0, a1);
#pragma unroll
      for (int off = 8; off >= 1; off >>= 1)
        rmax = fmaxf(rmax, __shfl_xor(rmax, off));
      const float mn    = fmaxf(mrow[r], rmax);
      const float alpha = __expf(mrow[r] - mn);
      mrow[r] = mn;
      const float e0 = __expf(a0 - mn);
      const float e1 = __expf(a1 - mn);
      float rsum = e0 + e1;
#pragma unroll
      for (int off = 8; off >= 1; off >>= 1)
        rsum += __shfl_xor(rsum, off);
      lrow[r] = lrow[r] * alpha + rsum;
#pragma unroll
      for (int j = 0; j < 4; ++j) o[j][r] *= alpha;
      // stage P (bf16) to this wave's LDS region in C-layout positions
      pl[mm * 32 + n]      = (__bf16)e0;
      pl[mm * 32 + 16 + n] = (__bf16)e1;
    }
    asm volatile("s_wait_dscnt 0x0" ::: "memory");  // P stores visible

    // ---- re-read P as an A-layout 16x32 fragment ----
    FragBF pf;
    pf.q[0] = *(const uint4*)&pl[n * 32 + abase];
    pf.q[1] = *(const uint4*)&pl[n * 32 + 8 + abase];

    // ---- O += P * V  (V fragments already resident in registers) ----
#pragma unroll
    for (int j = 0; j < 4; ++j)
      o[j] = wmma_bf16(pf, vf[j], o[j]);
    asm volatile("s_wait_dscnt 0x0" ::: "memory");  // WAR guard on pl
  }

  // ---- normalize and write fp32 output [B,T,C], C = h*64 + d ----
#pragma unroll
  for (int r = 0; r < 8; ++r) {
    const int mm  = lo ? r : r + 8;
    const int t   = q0 + mm;
    const float inv = 1.0f / lrow[r];
#pragma unroll
    for (int j = 0; j < 4; ++j)
      out[((size_t)(b * 2048 + t)) * 1024 + h * 64 + j * 16 + n] = o[j][r] * inv;
  }
}

// ---------------------------------------------------------------------------
extern "C" void kernel_launch(void* const* d_in, const int* in_sizes, int n_in,
                              void* d_out, int out_size, void* d_ws, size_t ws_size,
                              hipStream_t stream) {
  (void)in_sizes; (void)n_in; (void)out_size; (void)ws_size;
  const float* x  = (const float*)d_in[0];
  const float* wq = (const float*)d_in[1];
  float* out = (float*)d_out;

  const size_t elems = (size_t)4 * 16 * 2048 * 64;  // per-tensor bf16 elements
  __bf16* Qg = (__bf16*)d_ws;
  __bf16* Kg = Qg + elems;
  __bf16* Vt = Kg + elems;

  dim3 g1(3072 / 128, 8192 / 128);   // 24 x 64 blocks
  qkv_gemm_kernel<<<g1, dim3(256), 0, stream>>>(x, wq, Qg, Kg, Vt);

  dim3 g2(2048 / 64, 16, 4);         // q-tiles x heads x batch
  flash_attn_kernel<<<g2, dim3(128), 0, stream>>>(Qg, Kg, Vt, out);
}
